// Attention_61881888801122
// MI455X (gfx1250) — matrix operand order repaired and checked by reading, not compiled, not run on hardware
//
#include <hip/hip_runtime.h>
#include <cstdint>
#include <cstddef>

typedef __attribute__((ext_vector_type(16))) _Float16 v16h;
typedef __attribute__((ext_vector_type(8)))  float    v8f;

// async load-to-LDS builtin operand types: int4 in AS1 (global) / AS3 (LDS)
typedef int v4i32 __attribute__((vector_size(16)));
typedef __attribute__((address_space(1))) v4i32 gas_v4i32;
typedef __attribute__((address_space(3))) v4i32 las_v4i32;

constexpr int Bq = 4;
constexpr int Mq = 4096;
constexpr int Nq = 4096;
constexpr int Dq = 64;
constexpr int BM = 64;     // query rows per block (4 waves x 16 rows)
constexpr int BN = 64;     // KV tile
constexpr int NW = 4;      // waves per block
constexpr int NT = Nq / BN;
constexpr int LSTR = 72;   // padded LDS row stride in halves: 16B-aligned frags, 36*row injective mod 64 banks
constexpr int TSTR = 65;   // prepass transpose tile stride

// 0.125 (1/sqrt(64)) folded with log2(e): softmax in exp2 domain, one v_exp_f32 per element.
#define FA_C1 0.18033688011112042f

#if defined(__has_builtin)
#if __has_builtin(__builtin_amdgcn_global_load_async_to_lds_b128)
#define FA_ASYNC 1
#endif
#endif
#ifndef FA_ASYNC
#define FA_ASYNC 0
#endif

// ---------------- pre-pass: K -> f16 [b][n][d]; V -> f16 transposed [b][d][n] ----------------
__global__ __launch_bounds__(256) void prep_kernel(const float* __restrict__ k,
                                                   const float* __restrict__ v,
                                                   _Float16* __restrict__ kh,
                                                   _Float16* __restrict__ vt)
{
  __shared__ _Float16 Ls[Dq * TSTR];
  const int tid = threadIdx.x;
  const int blk = blockIdx.x;
  if (blk < Bq * (Nq / 64)) {
    // V transpose tile: 64 keys x 64 dims through LDS
    const int b  = blk / (Nq / 64);
    const int n0 = (blk % (Nq / 64)) * 64;
    const float* vb = v + ((size_t)b * Nq + n0) * Dq;
#pragma unroll
    for (int j = 0; j < 16; ++j) {
      int e = tid + 256 * j;
      int n = e >> 6, d = e & 63;
      Ls[d * TSTR + n] = (_Float16)vb[(size_t)n * Dq + d];   // coalesced f32 read
    }
    __syncthreads();
    _Float16* vtb = vt + (size_t)b * Dq * Nq + n0;
#pragma unroll
    for (int j = 0; j < 16; ++j) {
      int e = tid + 256 * j;
      int d = e >> 6, n = e & 63;
      vtb[(size_t)d * Nq + n] = Ls[d * TSTR + n];            // coalesced f16 write
    }
  } else {
    // K elementwise convert
    size_t base = (size_t)(blk - Bq * (Nq / 64)) * 4096;
#pragma unroll
    for (int j = 0; j < 16; ++j) {
      size_t e = base + (size_t)tid + 256 * j;
      kh[e] = (_Float16)k[e];
    }
  }
}

// ---------------- main flash-attention kernel ----------------
// Double-buffered staging: tile i+1 is fetched (async DMA to LDS when available)
// while tile i is computed; s_wait_asynccnt<=8 drains only the older batch.
template <bool PRE>
__global__ __launch_bounds__(128) void fattn_kernel(
    const float* __restrict__ q, const float* __restrict__ k,
    const float* __restrict__ v, const unsigned char* __restrict__ mask,
    const _Float16* __restrict__ kh, const _Float16* __restrict__ vt,
    float* __restrict__ out)
{
  __shared__ alignas(32) _Float16 Ks[2][BN * LSTR];     // K tile, [n][d] halves (double buffered)
  __shared__ alignas(32) _Float16 Vs[2][Dq * LSTR];     // V tile, transposed [d][n] halves
  __shared__ alignas(32) _Float16 Ps[NW * 16 * LSTR];   // per-wave P staging, [m][n] halves

  const int tid  = threadIdx.x;
  const int wave = tid >> 5;
  const int lane = tid & 31;
  const int hgrp = lane >> 4;    // lane half-group
  const int lcol = lane & 15;

  const int b  = blockIdx.x / (Mq / BM);
  const int m0 = (blockIdx.x % (Mq / BM)) * BM;
  const int mw = m0 + wave * 16;

  const float* Qb = q + ((size_t)b * Mq + mw) * Dq;
  const unsigned char* Mb = mask + (size_t)b * Mq * Nq;
  float* Ob = out + ((size_t)b * Mq + mw) * Dq;

  _Float16* Pw = &Ps[wave * 16 * LSTR];

  // stage tile starting at key n0s into LDS buffer bufs
  auto stage = [&](int n0s, int bufs) {
    if (PRE) {
#pragma unroll
      for (int jj = 0; jj < 4; ++jj) {
        int c   = tid + 128 * jj;         // 512 x 16B chunks each for K and V
        int row = c >> 3;                 // tile row (n for K, d for V)
        int cid = (c & 7) * 8;            // 16B chunk offset in halves
        const _Float16* gk = kh + ((size_t)b * Nq + n0s + row) * Dq + cid;
        const _Float16* gv = vt + ((size_t)b * Dq + row) * Nq + n0s + cid;
#if FA_ASYNC
        __builtin_amdgcn_global_load_async_to_lds_b128(
            (gas_v4i32*)gk, (las_v4i32*)&Ks[bufs][row * LSTR + cid], 0, 0);
        __builtin_amdgcn_global_load_async_to_lds_b128(
            (gas_v4i32*)gv, (las_v4i32*)&Vs[bufs][row * LSTR + cid], 0, 0);
#else
        *(uint4*)&Ks[bufs][row * LSTR + cid] = *(const uint4*)gk;
        *(uint4*)&Vs[bufs][row * LSTR + cid] = *(const uint4*)gv;
#endif
      }
    } else {
      // fallback: f32 -> f16 conversion staging (d_ws too small for pre-pass)
      const float* Kb = k + (size_t)b * Nq * Dq;
      const float* Vb = v + (size_t)b * Nq * Dq;
#pragma unroll
      for (int j = 0; j < 16; ++j) {
        int e  = tid + 128 * j;
        int n  = e >> 5;
        int d2 = (e & 31) * 2;
        const float* kp = Kb + (size_t)(n0s + n) * Dq + d2;
        Ks[bufs][n * LSTR + d2]       = (_Float16)kp[0];
        Ks[bufs][n * LSTR + d2 + 1]   = (_Float16)kp[1];
        const float* vp = Vb + (size_t)(n0s + n) * Dq + d2;
        Vs[bufs][(d2)     * LSTR + n] = (_Float16)vp[0];
        Vs[bufs][(d2 + 1) * LSTR + n] = (_Float16)vp[1];
      }
    }
  };

  // ---- Q A-fragments (16x32 f16 A layout), loaded once ----
  v16h aq[2];
#pragma unroll
  for (int i = 0; i < 2; ++i) {
    const float* qp = Qb + (size_t)lcol * Dq + 32 * i + 8 * hgrp;
#pragma unroll
    for (int j = 0; j < 8; ++j) aq[i][j]     = (_Float16)qp[j];
#pragma unroll
    for (int j = 0; j < 8; ++j) aq[i][8 + j] = (_Float16)qp[16 + j];
  }

  v8f acc[4];
  const v8f vzero = {};
#pragma unroll
  for (int i = 0; i < 4; ++i) acc[i] = vzero;
  float rm[8], rl[8];
#pragma unroll
  for (int r = 0; r < 8; ++r) { rm[r] = -1.0e30f; rl[r] = 0.0f; }

  stage(0, 0);   // prologue: prefetch tile 0

  for (int it = 0; it < NT; ++it) {
    const int n0  = it * BN;
    const int buf = it & 1;

    __syncthreads();   // all waves done reading buffer buf^1 (from iteration it-1)

    // prefetch next tile into the other buffer (last iter: harmless dummy refetch of tile 0)
    stage((it + 1 < NT) ? (n0 + BN) : 0, buf ^ 1);
#if FA_ASYNC
    if (PRE) asm volatile("s_wait_asynccnt 0x8" ::: "memory");  // drain tile-it batch only
#endif
    __syncthreads();   // tile it resident in LDS for all waves

    const _Float16* Ksb = Ks[buf];
    const _Float16* Vsb = Vs[buf];

    // ---- S = Q K^T (raw scores) for this wave's 16 rows x 64 keys ----
    v8f s[4];
#pragma unroll
    for (int ns = 0; ns < 4; ++ns) {
      v8f c = vzero;
#pragma unroll
      for (int i = 0; i < 2; ++i) {
        const _Float16* kp = &Ksb[(ns * 16 + lcol) * LSTR + 32 * i + 8 * hgrp];
        v16h kb;
#pragma unroll
        for (int j = 0; j < 8; ++j) kb[j] = kp[j];
#pragma unroll
        for (int j = 0; j < 8; ++j) kb[8 + j] = kp[16 + j];
        c = __builtin_amdgcn_wmma_f32_16x16x32_f16(false, aq[i], false, kb,
                                                   (short)0, c, false, false);
      }
      s[ns] = c;
    }

    // ---- mask (-inf), online softmax on raw scores; scale folded into exp2 constant ----
    float tmax[8];
#pragma unroll
    for (int r = 0; r < 8; ++r) tmax[r] = -1.0e30f;
#pragma unroll
    for (int ns = 0; ns < 4; ++ns) {
#pragma unroll
      for (int r = 0; r < 8; ++r) {
        float sc = s[ns][r];
        size_t midx = (size_t)(mw + r + 8 * hgrp) * Nq + (size_t)(n0 + ns * 16 + lcol);
        if (Mb[midx]) sc = -__builtin_huge_valf();
        s[ns][r] = sc;
        tmax[r] = fmaxf(tmax[r], sc);
      }
    }
#pragma unroll
    for (int r = 0; r < 8; ++r) {
#pragma unroll
      for (int off = 8; off >= 1; off >>= 1)
        tmax[r] = fmaxf(tmax[r], __shfl_xor(tmax[r], off, 32));
    }
    float alpha[8];
#pragma unroll
    for (int r = 0; r < 8; ++r) {
      float mnew = fmaxf(rm[r], tmax[r]);
      alpha[r] = exp2f((rm[r] - mnew) * FA_C1);
      rm[r] = mnew;
    }
    float tsum[8];
#pragma unroll
    for (int r = 0; r < 8; ++r) tsum[r] = 0.0f;
#pragma unroll
    for (int ns = 0; ns < 4; ++ns) {
#pragma unroll
      for (int r = 0; r < 8; ++r) {
        float p = exp2f((s[ns][r] - rm[r]) * FA_C1);   // masked -> exp2(-inf) = 0
        tsum[r] += p;
        Pw[(r + 8 * hgrp) * LSTR + ns * 16 + lcol] = (_Float16)p;
      }
    }
#pragma unroll
    for (int r = 0; r < 8; ++r) {
#pragma unroll
      for (int off = 8; off >= 1; off >>= 1)
        tsum[r] += __shfl_xor(tsum[r], off, 32);
      rl[r] = rl[r] * alpha[r] + tsum[r];
    }
#pragma unroll
    for (int d_ = 0; d_ < 4; ++d_) {
#pragma unroll
      for (int r = 0; r < 8; ++r) acc[d_][r] *= alpha[r];
    }

    // same-wave LDS RAW fence (P staged above, read back as A fragments below)
    asm volatile("s_wait_dscnt 0" ::: "memory");

    // ---- O += P V ----
#pragma unroll
    for (int nc = 0; nc < 2; ++nc) {
      const _Float16* pp = &Pw[lcol * LSTR + nc * 32 + 8 * hgrp];
      v16h ap;
#pragma unroll
      for (int j = 0; j < 8; ++j) ap[j]     = pp[j];
#pragma unroll
      for (int j = 0; j < 8; ++j) ap[8 + j] = pp[16 + j];
#pragma unroll
      for (int d_ = 0; d_ < 4; ++d_) {
        const _Float16* vp = &Vsb[(d_ * 16 + lcol) * LSTR + nc * 32 + 8 * hgrp];
        v16h vb;
#pragma unroll
        for (int j = 0; j < 8; ++j) vb[j] = vp[j];
#pragma unroll
        for (int j = 0; j < 8; ++j) vb[8 + j] = vp[16 + j];
        acc[d_] = __builtin_amdgcn_wmma_f32_16x16x32_f16(false, ap, false, vb,
                                                         (short)0, acc[d_], false, false);
      }
    }
  }

  // ---- normalize and store ----
#pragma unroll
  for (int r = 0; r < 8; ++r) {
    float inv = 1.0f / rl[r];
#pragma unroll
    for (int d_ = 0; d_ < 4; ++d_)
      Ob[(size_t)(r + 8 * hgrp) * Dq + d_ * 16 + lcol] = acc[d_][r] * inv;
  }
}

extern "C" void kernel_launch(void* const* d_in, const int* in_sizes, int n_in,
                              void* d_out, int out_size, void* d_ws, size_t ws_size,
                              hipStream_t stream) {
  (void)in_sizes; (void)n_in; (void)out_size;
  const float* q = (const float*)d_in[0];
  const float* k = (const float*)d_in[1];
  const float* v = (const float*)d_in[2];
  const unsigned char* mask = (const unsigned char*)d_in[3];  // jax bool = 1 byte
  float* o = (float*)d_out;

  const size_t elems = (size_t)Bq * Nq * Dq;                  // 1M per tensor
  const size_t need  = 2 * elems * sizeof(_Float16);          // kh + vt = 4 MB
  if (ws_size >= need) {
    _Float16* kh = (_Float16*)d_ws;
    _Float16* vt = kh + elems;
    prep_kernel<<<dim3(512), dim3(256), 0, stream>>>(k, v, kh, vt);
    fattn_kernel<true><<<dim3(Bq * (Mq / BM)), dim3(128), 0, stream>>>(
        q, k, v, mask, kh, vt, o);
  } else {
    fattn_kernel<false><<<dim3(Bq * (Mq / BM)), dim3(128), 0, stream>>>(
        q, k, v, mask, nullptr, nullptr, o);
  }
}
